// TransformerBlock_37503654428853
// MI455X (gfx1250) — matrix operand order repaired and checked by reading, not compiled, not run on hardware
//
#include <hip/hip_runtime.h>
#include <hip/hip_bf16.h>

// ---------------------------------------------------------------------------
// Types for CDNA5 WMMA (wave32): v_wmma_f32_16x16x32_bf16
// ---------------------------------------------------------------------------
typedef __bf16 bf16;
typedef __attribute__((ext_vector_type(8)))  __bf16 v8bf;
typedef __attribute__((ext_vector_type(16))) __bf16 v16bf;
typedef __attribute__((ext_vector_type(8)))  float  v8f;

union BF16x16 { v16bf v; v8bf h[2]; };

__device__ __forceinline__ v8f wmma_bf16(v16bf a, v16bf b, v8f c) {
    // 8 args: (neg_a, A, neg_b, B, c_mod, C, reuse_a, reuse_b)
    return __builtin_amdgcn_wmma_f32_16x16x32_bf16(
        false, a, false, b, (short)0, c, false, false);
}

// A-fragment (16x32 bf16, row-major source, leading dim `ld` elements).
// lane: row = lane&15, half = lane>>4. elems 0..7: K=half*8+e ; 8..15: K=16+half*8+(e-8)
__device__ __forceinline__ v16bf load_a_frag(const bf16* base, int ld) {
    const int lane = threadIdx.x & 31;
    const int row  = lane & 15;
    const int half = lane >> 4;
    const bf16* p = base + (size_t)row * ld + half * 8;
    BF16x16 u;
    u.h[0] = *(const v8bf*)(p);
    u.h[1] = *(const v8bf*)(p + 16);
    return u.v;
}

// B-fragment (32x16 bf16). Source stored "transposed": row = output column N,
// contiguous K. lane: n = lane&15, elems 0..7: K=half*8+e ; 8..15: K=16+half*8+(e-8)
__device__ __forceinline__ v16bf load_b_frag(const bf16* base, int ld) {
    const int lane = threadIdx.x & 31;
    const int n    = lane & 15;
    const int half = lane >> 4;
    const bf16* p = base + (size_t)n * ld + half * 8;
    BF16x16 u;
    u.h[0] = *(const v8bf*)(p);
    u.h[1] = *(const v8bf*)(p + 16);
    return u.v;
}

// Async global->LDS copy of 16B per lane (GLOBAL_LOAD_ASYNC_TO_LDS_B128,
// tracked by ASYNCcnt). ldsoff = LDS byte offset, gaddr = global address.
__device__ __forceinline__ void async_copy_b128(unsigned ldsoff,
                                                unsigned long long gaddr) {
    asm volatile("global_load_async_to_lds_b128 %0, %1, off"
                 :: "v"(ldsoff), "v"(gaddr) : "memory");
}
__device__ __forceinline__ void wait_asynccnt0() {
    asm volatile("s_wait_asynccnt 0x0" ::: "memory");
}

// ---------------------------------------------------------------------------
// Weight convert + transpose: W fp32 [K,N] -> Wt bf16 [N,K]
// ---------------------------------------------------------------------------
__global__ __launch_bounds__(256)
void convert_transpose_kernel(const float* __restrict__ W,
                              bf16* __restrict__ Wt, int K, int N) {
    long idx = (long)blockIdx.x * blockDim.x + threadIdx.x;
    if (idx >= (long)K * N) return;
    int n = (int)(idx / K);
    int k = (int)(idx % K);
    Wt[idx] = (bf16)W[(size_t)k * N + n];
}

// ---------------------------------------------------------------------------
// LayerNorm (D=1024), one block (256 thr) per row, bf16 output
// ---------------------------------------------------------------------------
__global__ __launch_bounds__(256)
void layernorm_kernel(const float* __restrict__ x,
                      const float* __restrict__ g,
                      const float* __restrict__ be,
                      bf16* __restrict__ out) {
    const int D = 1024;
    const int row = blockIdx.x, t = threadIdx.x;
    const float4 v = ((const float4*)(x + (size_t)row * D))[t];
    float s  = v.x + v.y + v.z + v.w;
    float s2 = v.x * v.x + v.y * v.y + v.z * v.z + v.w * v.w;
#pragma unroll
    for (int off = 16; off >= 1; off >>= 1) {
        s  += __shfl_xor(s,  off, 32);
        s2 += __shfl_xor(s2, off, 32);
    }
    __shared__ float red[16];
    const int wid = t >> 5, lane = t & 31;
    if (lane == 0) { red[wid] = s; red[8 + wid] = s2; }
    __syncthreads();
    float tot = 0.f, tot2 = 0.f;
#pragma unroll
    for (int i = 0; i < 8; ++i) { tot += red[i]; tot2 += red[8 + i]; }
    const float mu  = tot * (1.0f / D);
    const float var = tot2 * (1.0f / D) - mu * mu;
    const float rs  = rsqrtf(var + 1e-5f);
    const int c0 = t * 4;
    bf16* o = out + (size_t)row * D + c0;
    o[0] = (bf16)((v.x - mu) * rs * g[c0 + 0] + be[c0 + 0]);
    o[1] = (bf16)((v.y - mu) * rs * g[c0 + 1] + be[c0 + 1]);
    o[2] = (bf16)((v.z - mu) * rs * g[c0 + 2] + be[c0 + 2]);
    o[3] = (bf16)((v.w - mu) * rs * g[c0 + 3] + be[c0 + 3]);
}

// ---------------------------------------------------------------------------
// WMMA GEMM: C[M,N] = epilogue(A[M,K] @ B[K,N])
//   A bf16 row-major [M,K]; Bt bf16 = B transposed, [N,K]
//   Block 128 thr (4 waves), tile 64x64, K staged 64 at a time via
//   async global->LDS copies; waves 2x2, each 32x32, 8 WMMAs per stage.
//   EPI 0: bf16 out                1: bias + ReLU, bf16 out
//   EPI 2: + residual, f32 out     3: bias + residual, f32 out
// ---------------------------------------------------------------------------
template <int EPI>
__global__ __launch_bounds__(128)
void gemm_bf16_kernel(const bf16* __restrict__ A,
                      const bf16* __restrict__ Bt,
                      void* __restrict__ Cout,
                      const float* __restrict__ bias,
                      const float* __restrict__ resid,
                      int M, int N, int K) {
    constexpr int LD = 72;  // 144B rows -> 16B aligned v8bf accesses
    __shared__ bf16 As[64 * LD];
    __shared__ bf16 Bs[64 * LD];
    const int t = threadIdx.x;
    const int w = t >> 5, lane = t & 31;
    const int wm = (w >> 1) * 32, wn = (w & 1) * 32;
    const int m0 = blockIdx.y * 64, n0 = blockIdx.x * 64;
    const int srow = t >> 1, soff = (t & 1) * 32;   // each thread: 32 elems (64B)
    const unsigned ldsA = (unsigned)(size_t)&As[srow * LD + soff];
    const unsigned ldsB = (unsigned)(size_t)&Bs[srow * LD + soff];
    v8f acc[2][2] = {};
    for (int k0 = 0; k0 < K; k0 += 64) {
        const bf16* pa = A  + (size_t)(m0 + srow) * K + k0 + soff;
        const bf16* pb = Bt + (size_t)(n0 + srow) * K + k0 + soff;
        __syncthreads();
        // Stage the 64x64 A / B^T tiles with async global->LDS b128 copies.
        {
            const unsigned long long ga = (unsigned long long)(size_t)pa;
            const unsigned long long gb = (unsigned long long)(size_t)pb;
#pragma unroll
            for (int j = 0; j < 4; ++j) {
                async_copy_b128(ldsA + 16u * j, ga + 16ull * j);
                async_copy_b128(ldsB + 16u * j, gb + 16ull * j);
            }
        }
        if (k0 + 64 < K) {  // prefetch next K-tile (global_prefetch_b8)
            __builtin_prefetch(pa + 64, 0, 2);
            __builtin_prefetch(pb + 64, 0, 2);
        }
        wait_asynccnt0();   // this wave's async copies landed in LDS
        __syncthreads();    // all waves' copies visible
#pragma unroll
        for (int ks = 0; ks < 2; ++ks) {
            v16bf a0 = load_a_frag(&As[(wm + 0)  * LD + ks * 32], LD);
            v16bf a1 = load_a_frag(&As[(wm + 16) * LD + ks * 32], LD);
            v16bf b0 = load_b_frag(&Bs[(wn + 0)  * LD + ks * 32], LD);
            v16bf b1 = load_b_frag(&Bs[(wn + 16) * LD + ks * 32], LD);
            acc[0][0] = wmma_bf16(a0, b0, acc[0][0]);
            acc[0][1] = wmma_bf16(a0, b1, acc[0][1]);
            acc[1][0] = wmma_bf16(a1, b0, acc[1][0]);
            acc[1][1] = wmma_bf16(a1, b1, acc[1][1]);
        }
    }
    // Epilogue: C layout — col = lane&15, row = r + 8*(lane>>4)
    const int half = lane >> 4, nl = lane & 15;
#pragma unroll
    for (int i = 0; i < 2; ++i)
#pragma unroll
        for (int j = 0; j < 2; ++j)
#pragma unroll
            for (int r = 0; r < 8; ++r) {
                const int row = m0 + wm + i * 16 + r + 8 * half;
                const int col = n0 + wn + j * 16 + nl;
                float v = acc[i][j][r];
                if (EPI == 1 || EPI == 3) v += bias[col];
                if (EPI == 1) v = fmaxf(v, 0.0f);
                if (EPI == 2 || EPI == 3) v += resid[(size_t)row * N + col];
                if (EPI <= 1) ((bf16*)Cout)[(size_t)row * N + col] = (bf16)v;
                else          ((float*)Cout)[(size_t)row * N + col] = v;
            }
}

// ---------------------------------------------------------------------------
// Flash attention (causal), Q/K/V bf16 packed [B*S, 1024] = [.., h*64+d]
// Block = (b, h, 64-query chunk); 4 waves x 16 query rows.
// ---------------------------------------------------------------------------
__global__ __launch_bounds__(128)
void attention_kernel(const bf16* __restrict__ Q,
                      const bf16* __restrict__ Kb,
                      const bf16* __restrict__ Vb,
                      bf16* __restrict__ Ob) {
    constexpr int S = 512, DK = 64, DM = 1024, QB = 64, VT_LD = 40;
    __shared__ bf16 Vt[DK * VT_LD];     // V chunk transposed: [64 d][32 keys]+pad
    __shared__ bf16 Plds[4 * 16 * 32];  // per-wave P tile 16x32
    const int t = threadIdx.x, w = t >> 5, lane = t & 31;
    const int half = lane >> 4, nl = lane & 15;
    const int blk = blockIdx.x;
    const int qb = blk & 7;
    const int h  = (blk >> 3) & 15;
    const int b  = blk >> 7;
    const int q0w = qb * QB + w * 16;

    const size_t rowQ = (size_t)(b * S + q0w) * DM + h * DK;
    const v16bf aQ0 = load_a_frag(Q + rowQ, DM);        // d = 0..31
    const v16bf aQ1 = load_a_frag(Q + rowQ + 32, DM);   // d = 32..63

    float mrow[8], lrow[8];
    v8f O[4] = {};
#pragma unroll
    for (int r = 0; r < 8; ++r) { mrow[r] = -3.0e38f; lrow[r] = 0.0f; }

    const int nchunk = 2 * (qb + 1);       // keys 0 .. 64*(qb+1), chunks of 32
    const int vkey = t >> 2, vdp = (t & 3) * 16;
    bf16* Pw = Plds + w * (16 * 32);

    for (int c = 0; c < nchunk; ++c) {
        const int k0 = c * 32;
        __syncthreads();  // previous chunk's Vt/P reads complete
        // Stage V chunk transposed into LDS: Vt[d][key]
        {
            const bf16* vp = Vb + (size_t)(b * S + k0 + vkey) * DM + h * DK + vdp;
            v8bf v0 = *(const v8bf*)vp, v1 = *(const v8bf*)(vp + 8);
#pragma unroll
            for (int i = 0; i < 8; ++i) {
                Vt[(vdp + i)     * VT_LD + vkey] = v0[i];
                Vt[(vdp + 8 + i) * VT_LD + vkey] = v1[i];
            }
        }
        // Scores: S = Q @ K^T  (d contiguous in K rows -> direct global B frags)
        const bf16* kbase = Kb + (size_t)(b * S + k0) * DM + h * DK;
        v16bf bK00 = load_b_frag(kbase, DM);                       // keys 0..15, d 0..31
        v16bf bK01 = load_b_frag(kbase + 32, DM);                  // keys 0..15, d 32..63
        v16bf bK10 = load_b_frag(kbase + (size_t)16 * DM, DM);     // keys 16..31
        v16bf bK11 = load_b_frag(kbase + (size_t)16 * DM + 32, DM);
        v8f s0 = {}, s1 = {};
        s0 = wmma_bf16(aQ0, bK00, s0); s0 = wmma_bf16(aQ1, bK01, s0);
        s1 = wmma_bf16(aQ0, bK10, s1); s1 = wmma_bf16(aQ1, bK11, s1);
        // Online softmax over the 32 new columns
#pragma unroll
        for (int r = 0; r < 8; ++r) {
            const int q = q0w + r + 8 * half;
            float x0 = (float)s0[r] * 0.125f;
            float x1 = (float)s1[r] * 0.125f;
            if (k0 + nl      > q) x0 = -1.0e30f;   // causal mask
            if (k0 + 16 + nl > q) x1 = -1.0e30f;
            float tmax = fmaxf(x0, x1);
            tmax = fmaxf(tmax, __shfl_xor(tmax, 1, 32));
            tmax = fmaxf(tmax, __shfl_xor(tmax, 2, 32));
            tmax = fmaxf(tmax, __shfl_xor(tmax, 4, 32));
            tmax = fmaxf(tmax, __shfl_xor(tmax, 8, 32));
            const float mnew = fmaxf(mrow[r], tmax);
            const float p0 = __expf(x0 - mnew);
            const float p1 = __expf(x1 - mnew);
            float ps = p0 + p1;
            ps += __shfl_xor(ps, 1, 32);
            ps += __shfl_xor(ps, 2, 32);
            ps += __shfl_xor(ps, 4, 32);
            ps += __shfl_xor(ps, 8, 32);
            const float corr = __expf(mrow[r] - mnew);
            mrow[r] = mnew;
            lrow[r] = lrow[r] * corr + ps;
#pragma unroll
            for (int dt = 0; dt < 4; ++dt) O[dt][r] *= corr;
            Pw[(r + 8 * half) * 32 + nl]      = (bf16)p0;   // repack C->A via LDS
            Pw[(r + 8 * half) * 32 + 16 + nl] = (bf16)p1;
        }
        __syncthreads();  // Vt staged + P written
        const v16bf aP = load_a_frag(Pw, 32);               // 16x32, K = 32 keys
#pragma unroll
        for (int dt = 0; dt < 4; ++dt) {
            v16bf bV = load_b_frag(&Vt[dt * 16 * VT_LD], VT_LD);
            O[dt] = wmma_bf16(aP, bV, O[dt]);
        }
    }
    // Normalize and store
#pragma unroll
    for (int r = 0; r < 8; ++r) {
        const float inv = 1.0f / lrow[r];
        const size_t orow = (size_t)(b * S + q0w + r + 8 * half) * DM + h * DK;
#pragma unroll
        for (int dt = 0; dt < 4; ++dt)
            Ob[orow + dt * 16 + nl] = (bf16)(O[dt][r] * inv);
    }
}

// ---------------------------------------------------------------------------
// Host-side orchestration
// ---------------------------------------------------------------------------
extern "C" void kernel_launch(void* const* d_in, const int* in_sizes, int n_in,
                              void* d_out, int out_size, void* d_ws, size_t ws_size,
                              hipStream_t stream) {
    constexpr int B = 16, S = 512, DM = 1024, DFF = 4096;
    constexpr int ROWS = B * S;  // 8192

    const float* x   = (const float*)d_in[0];
    const float* Wq  = (const float*)d_in[1];
    const float* Wk  = (const float*)d_in[2];
    const float* Wv  = (const float*)d_in[3];
    const float* Wo  = (const float*)d_in[4];
    const float* W1  = (const float*)d_in[5];
    const float* b1  = (const float*)d_in[6];
    const float* W2  = (const float*)d_in[7];
    const float* b2  = (const float*)d_in[8];
    const float* g1  = (const float*)d_in[9];
    const float* be1 = (const float*)d_in[10];
    const float* g2  = (const float*)d_in[11];
    const float* be2 = (const float*)d_in[12];
    float* out = (float*)d_out;

    char* ws = (char*)d_ws;
    auto alloc = [&](size_t bytes) {
        char* p = ws;
        ws += (bytes + 255) & ~(size_t)255;
        return p;
    };
    bf16* WqT = (bf16*)alloc((size_t)DM * DM * 2);
    bf16* WkT = (bf16*)alloc((size_t)DM * DM * 2);
    bf16* WvT = (bf16*)alloc((size_t)DM * DM * 2);
    bf16* WoT = (bf16*)alloc((size_t)DM * DM * 2);
    bf16* W1T = (bf16*)alloc((size_t)DM * DFF * 2);
    bf16* W2T = (bf16*)alloc((size_t)DFF * DM * 2);
    bf16* h1  = (bf16*)alloc((size_t)ROWS * DM * 2);
    bf16* Qb  = (bf16*)alloc((size_t)ROWS * DM * 2);
    bf16* Kb  = (bf16*)alloc((size_t)ROWS * DM * 2);
    bf16* Vb  = (bf16*)alloc((size_t)ROWS * DM * 2);
    bf16* At  = (bf16*)alloc((size_t)ROWS * DM * 2);
    float* x1 = (float*)alloc((size_t)ROWS * DM * 4);
    bf16* h2  = (bf16*)alloc((size_t)ROWS * DM * 2);
    bf16* ff  = (bf16*)alloc((size_t)ROWS * DFF * 2);

    // 1) convert + transpose weights to bf16 [N,K]
    {
        int blk = 256;
        int g1n = (DM * DM) / blk, g2n = (DM * DFF) / blk;
        convert_transpose_kernel<<<g1n, blk, 0, stream>>>(Wq, WqT, DM, DM);
        convert_transpose_kernel<<<g1n, blk, 0, stream>>>(Wk, WkT, DM, DM);
        convert_transpose_kernel<<<g1n, blk, 0, stream>>>(Wv, WvT, DM, DM);
        convert_transpose_kernel<<<g1n, blk, 0, stream>>>(Wo, WoT, DM, DM);
        convert_transpose_kernel<<<g2n, blk, 0, stream>>>(W1, W1T, DM, DFF);
        convert_transpose_kernel<<<g2n, blk, 0, stream>>>(W2, W2T, DFF, DM);
    }
    // 2) h1 = LN1(x) -> bf16
    layernorm_kernel<<<ROWS, 256, 0, stream>>>(x, g1, be1, h1);
    // 3) Q, K, V projections
    dim3 gqkv(DM / 64, ROWS / 64);
    gemm_bf16_kernel<0><<<gqkv, 128, 0, stream>>>(h1, WqT, Qb, nullptr, nullptr, ROWS, DM, DM);
    gemm_bf16_kernel<0><<<gqkv, 128, 0, stream>>>(h1, WkT, Kb, nullptr, nullptr, ROWS, DM, DM);
    gemm_bf16_kernel<0><<<gqkv, 128, 0, stream>>>(h1, WvT, Vb, nullptr, nullptr, ROWS, DM, DM);
    // 4) causal flash attention
    attention_kernel<<<B * 16 * (S / 64), 128, 0, stream>>>(Qb, Kb, Vb, At);
    // 5) x1 = x + attn @ Wo
    gemm_bf16_kernel<2><<<gqkv, 128, 0, stream>>>(At, WoT, x1, nullptr, x, ROWS, DM, DM);
    // 6) h2 = LN2(x1)
    layernorm_kernel<<<ROWS, 256, 0, stream>>>(x1, g2, be2, h2);
    // 7) ff = relu(h2 @ W1 + b1)
    dim3 gff1(DFF / 64, ROWS / 64);
    gemm_bf16_kernel<1><<<gff1, 128, 0, stream>>>(h2, W1T, ff, b1, nullptr, ROWS, DFF, DM);
    // 8) out = x1 + ff @ W2 + b2
    gemm_bf16_kernel<3><<<gqkv, 128, 0, stream>>>(ff, W2T, out, b2, x1, ROWS, DM, DFF);
}
